// GCN_5995774345732
// MI455X (gfx1250) — matrix operand order repaired and checked by reading, not compiled, not run on hardware
//
#include <hip/hip_runtime.h>

typedef __attribute__((ext_vector_type(16))) _Float16 v16h;
typedef __attribute__((ext_vector_type(8)))  float    v8f;

#define NFEAT  256
#define NHID    64
#define NCLASS  40

// ---------------- utility kernels ----------------

__global__ void zero_kernel(float* __restrict__ p, int count) {
  int i = blockIdx.x * blockDim.x + threadIdx.x;
  if (i < count) p[i] = 0.0f;
}

__global__ void degree_kernel(const int* __restrict__ dst, float* __restrict__ deg, int E) {
  int e = blockIdx.x * blockDim.x + threadIdx.x;
  if (e < E) atomicAdd(&deg[dst[e]], 1.0f);
}

__global__ void dinv_kernel(float* __restrict__ deg, int n) {
  int i = blockIdx.x * blockDim.x + threadIdx.x;
  if (i < n) deg[i] = rsqrtf(deg[i] + 1.0f);   // +1 self-loop; always > 0
}

// per-edge symmetric norm: norm[e] = dinv[src]*dinv[dst]
__global__ void norm_kernel(const int* __restrict__ src, const int* __restrict__ dst,
                            const float* __restrict__ dinv, float* __restrict__ norm, int E) {
  int e = blockIdx.x * blockDim.x + threadIdx.x;
  if (e < E) norm[e] = dinv[src[e]] * dinv[dst[e]];
}

// ---------------- WMMA GEMM: C[M x NOUT] = A[M x K] * B[K x NOUT] ----------------
// One wave32 per 16x16 output tile. B panel (K x 16 cols of this blockIdx.y) is
// staged once per block into LDS, pre-swizzled into WMMA B-fragment order
// (branchless zero-fill for padded columns), then each wave pulls 32B fragments
// via ds loads. A is fetched as float4 runs and packed f32->f16 in-register.

__device__ inline void pack8(v16h& a, int base, float4 lo, float4 hi) {
  a[base + 0] = (_Float16)lo.x; a[base + 1] = (_Float16)lo.y;
  a[base + 2] = (_Float16)lo.z; a[base + 3] = (_Float16)lo.w;
  a[base + 4] = (_Float16)hi.x; a[base + 5] = (_Float16)hi.y;
  a[base + 6] = (_Float16)hi.z; a[base + 7] = (_Float16)hi.w;
}

template<int K, int NOUT>
__global__ void gemm_wmma_kernel(const float* __restrict__ A,
                                 const float* __restrict__ B,
                                 float* __restrict__ C, int M) {
  constexpr int KB = K / 32;                 // number of 32-deep K blocks
  __shared__ _Float16 bsmem[KB * 32 * 16];   // [kb][lane][16 halfs] fragment order

  const int tid   = threadIdx.x;
  const int lane  = tid & 31;
  const int wave  = tid >> 5;
  const int hs    = lane >> 4;               // half-wave select
  const int lm    = lane & 15;
  const int tileN = blockIdx.y;

  // --- cooperative, branchless stage of the B panel into LDS ---
  // element idx -> kb = idx>>9, ln = (idx>>4)&31, j = idx&15
  for (int idx = tid; idx < KB * 512; idx += blockDim.x) {
    int kb = idx >> 9;
    int ln = (idx >> 4) & 31;
    int j  = idx & 15;
    int k  = kb * 32 + (ln >> 4) * 8 + (j & 7) + (j >> 3) * 16;   // B-fragment k for lane ln
    int col  = tileN * 16 + (ln & 15);
    int colc = col < NOUT ? col : NOUT - 1;  // clamp address (no divergent load)
    float bv = B[(size_t)k * NOUT + colc];
    bsmem[idx] = (_Float16)(col < NOUT ? bv : 0.0f);   // v_cndmask, branchless
  }
  __syncthreads();

  const int tileM  = blockIdx.x * (blockDim.x >> 5) + wave;
  const int Mtiles = (M + 15) >> 4;
  if (tileM >= Mtiles) return;               // wave-uniform; no barriers follow

  const int row = tileM * 16 + lm;
  const int rA  = row < M ? row : M - 1;     // clamp (stores are guarded)
  const int col = tileN * 16 + lm;
  const bool colOK = (col < NOUT);

  const float* arow = A + (size_t)rA * K;
  const v16h* bfrag = reinterpret_cast<const v16h*>(bsmem);

  v8f c = {};
#pragma unroll
  for (int kb = 0; kb < KB; ++kb) {
    // A fragment: halfs j=0..7 -> k = kb*32 + hs*8 + j ; j=8..15 -> +16
    const int k0 = kb * 32 + hs * 8;
    float4 a0 = *reinterpret_cast<const float4*>(arow + k0);
    float4 a1 = *reinterpret_cast<const float4*>(arow + k0 + 4);
    float4 a2 = *reinterpret_cast<const float4*>(arow + k0 + 16);
    float4 a3 = *reinterpret_cast<const float4*>(arow + k0 + 20);
    v16h a;
    pack8(a, 0, a0, a1);
    pack8(a, 8, a2, a3);
    v16h b = bfrag[kb * 32 + lane];          // 32B LDS fragment load
    c = __builtin_amdgcn_wmma_f32_16x16x32_f16(
        /*neg_a=*/false, a, /*neg_b=*/false, b,
        /*c_mod=*/(short)0, c, /*reuse_a=*/false, /*reuse_b=*/false);
  }
  // C/D layout: lane lm = column, VGPR r = row r + hs*8
#pragma unroll
  for (int r = 0; r < 8; ++r) {
    int m = tileM * 16 + r + hs * 8;
    if (m < M && colOK) C[(size_t)m * NOUT + col] = c[r];
  }
}

// ---------------- edge aggregation: agg[dst] += h[src] * norm[e] ----------------
// Each thread owns QPT float4 quads of one edge: one src/dst/norm load amortized
// over 4*QPT features; gathers/atomics stay resident in the 192MB L2.

template<int F, int QPT>
__global__ void aggregate_kernel(const float* __restrict__ h,
                                 const int* __restrict__ src,
                                 const int* __restrict__ dst,
                                 const float* __restrict__ norm,
                                 float* __restrict__ agg, int E) {
  constexpr int TPE = F / (4 * QPT);         // threads per edge
  unsigned gid = blockIdx.x * blockDim.x + threadIdx.x;
  if (gid >= (unsigned)E * TPE) return;
  int e = gid / TPE;
  int t = gid - e * TPE;
  int s = src[e], d = dst[e];
  float nm = norm[e];
  const float4* hrow = reinterpret_cast<const float4*>(h + (size_t)s * F) + t * QPT;
  float* arow = agg + (size_t)d * F + t * QPT * 4;
#pragma unroll
  for (int q = 0; q < QPT; ++q) {
    float4 hv = hrow[q];
    atomicAdd(arow + q * 4 + 0, hv.x * nm);
    atomicAdd(arow + q * 4 + 1, hv.y * nm);
    atomicAdd(arow + q * 4 + 2, hv.z * nm);
    atomicAdd(arow + q * 4 + 3, hv.w * nm);
  }
}

// ---------------- epilogues ----------------

// agg = relu(agg + h1 * dinv^2 + b1)   (F = 64)
__global__ void finish1_kernel(float* __restrict__ agg, const float* __restrict__ h1,
                               const float* __restrict__ dinv, const float* __restrict__ b1,
                               int total) {
  int gid = blockIdx.x * blockDim.x + threadIdx.x;
  if (gid >= total) return;
  int i = gid >> 6;          // / NHID
  int f = gid & 63;          // % NHID
  float di = dinv[i];
  float v = agg[gid] + h1[gid] * di * di + b1[f];
  agg[gid] = v > 0.0f ? v : 0.0f;
}

// out = log_softmax(agg + out * dinv^2 + b2) per row of NCLASS
__global__ void finish2_kernel(const float* __restrict__ agg, float* __restrict__ out,
                               const float* __restrict__ dinv, const float* __restrict__ b2,
                               int n) {
  int i = blockIdx.x * blockDim.x + threadIdx.x;
  if (i >= n) return;
  float di = dinv[i];
  float d2 = di * di;
  size_t base = (size_t)i * NCLASS;
  float v[NCLASS];
  float mx = -3.4e38f;
#pragma unroll
  for (int f = 0; f < NCLASS; ++f) {
    float t = agg[base + f] + out[base + f] * d2 + b2[f];
    v[f] = t;
    mx = fmaxf(mx, t);
  }
  float s = 0.0f;
#pragma unroll
  for (int f = 0; f < NCLASS; ++f) s += __expf(v[f] - mx);
  float ls = __logf(s);
#pragma unroll
  for (int f = 0; f < NCLASS; ++f) out[base + f] = v[f] - mx - ls;
}

// ---------------- host launcher ----------------

extern "C" void kernel_launch(void* const* d_in, const int* in_sizes, int n_in,
                              void* d_out, int out_size, void* d_ws, size_t ws_size,
                              hipStream_t stream) {
  const float* x  = (const float*)d_in[0];
  const int*   ei = (const int*)d_in[1];
  const float* W1 = (const float*)d_in[2];
  const float* b1 = (const float*)d_in[3];
  const float* W2 = (const float*)d_in[4];
  const float* b2 = (const float*)d_in[5];

  const int N = in_sizes[0] / NFEAT;   // 100000
  const int E = in_sizes[1] / 2;       // 1600000
  const int* src = ei;
  const int* dst = ei + E;

  // workspace layout (floats): dinv | norm | h1 | agg1 | agg2   (~75 MB)
  float* ws   = (float*)d_ws;
  float* dinv = ws;
  size_t off  = 262144;                           // 1 MB reserved for dinv
  float* norm = ws + off; off += (size_t)E;
  float* h1   = ws + off; off += (size_t)N * NHID;
  float* agg1 = ws + off; off += (size_t)N * NHID;
  float* agg2 = ws + off;
  float* h2   = (float*)d_out;                    // reuse output buffer for h2

  const int B = 256;

  // degrees -> dinv = rsqrt(deg+1) -> per-edge norm
  zero_kernel<<<(N + B - 1) / B, B, 0, stream>>>(dinv, N);
  degree_kernel<<<(E + B - 1) / B, B, 0, stream>>>(dst, dinv, E);
  dinv_kernel<<<(N + B - 1) / B, B, 0, stream>>>(dinv, N);
  norm_kernel<<<(E + B - 1) / B, B, 0, stream>>>(src, dst, dinv, norm, E);

  // layer 1: h1 = x @ W1 (WMMA, LDS-staged B)
  {
    int Mtiles = (N + 15) / 16;
    dim3 grid((Mtiles + 7) / 8, NHID / 16);
    gemm_wmma_kernel<NFEAT, NHID><<<grid, 256, 0, stream>>>(x, W1, h1, N);
  }
  const int tot1 = N * NHID;
  zero_kernel<<<(tot1 + B - 1) / B, B, 0, stream>>>(agg1, tot1);
  {
    unsigned work = (unsigned)E * (NHID / 16);     // TPE = 64/(4*4) = 4
    aggregate_kernel<NHID, 4><<<(work + B - 1) / B, B, 0, stream>>>(h1, src, dst, norm, agg1, E);
  }
  finish1_kernel<<<(tot1 + B - 1) / B, B, 0, stream>>>(agg1, h1, dinv, b1, tot1);

  // layer 2: h2 = relu_h @ W2 (WMMA), N padded 40 -> 48 with guards
  {
    int Mtiles = (N + 15) / 16;
    dim3 grid((Mtiles + 7) / 8, (NCLASS + 15) / 16);
    gemm_wmma_kernel<NHID, NCLASS><<<grid, 256, 0, stream>>>(agg1, W2, h2, N);
  }
  const int tot2 = N * NCLASS;
  zero_kernel<<<(tot2 + B - 1) / B, B, 0, stream>>>(agg2, tot2);
  {
    unsigned work = (unsigned)E * 2;               // TPE = 40/(4*5) = 2
    aggregate_kernel<NCLASS, 5><<<(work + B - 1) / B, B, 0, stream>>>(h2, src, dst, norm, agg2, E);
  }
  finish2_kernel<<<(N + B - 1) / B, B, 0, stream>>>(agg2, h2, dinv, b2, N);
}
